// GraphFilter_26628797235681
// MI455X (gfx1250) — matrix operand order repaired and checked by reading, not compiled, not run on hardware
//
#include <hip/hip_runtime.h>

typedef __bf16 bf16_t;
typedef __attribute__((ext_vector_type(16))) __bf16 v16bf;
typedef __attribute__((ext_vector_type(8)))  __bf16 v8bf;
typedef __attribute__((ext_vector_type(4)))  __bf16 v4bf;
typedef __attribute__((ext_vector_type(8)))  float  v8f;

namespace gf {
constexpr int Tdim = 64, FIN = 64, FOUT = 64, Ndim = 128, Edim = 2, Ktap = 4;
constexpr int HROW = Edim * Ktap * FIN;   // 512, flattened H row length

// LDS strides in bf16 elements. All multiples of 8 -> 16B-aligned rows for b128.
constexpr int SA  = 136;  // row-major [64][136]  (A operands)
constexpr int SBT = 72;   // transposed tap [128][72]   (B operands, cols = f)
constexpr int SST = 136;  // transposed S  [128][136]   (B operands, cols = n')

// LDS offsets (bf16 elements)
constexpr int X0T = 0;                       // x[t]   transposed [128][72]
constexpr int X1o = X0T + Ndim * SBT;        // x[t-1] row-major  [64][136]
constexpr int X2o = X1o + FIN * SA;          // x[t-2]
constexpr int X3o = X2o + FIN * SA;          // x[t-3]
constexpr int ST0 = X3o + FIN * SA;          // S[t]^T   [128][136]
constexpr int ST1 = ST0 + Ndim * SST;        // S[t-1]^T
constexpr int ST2 = ST1 + Ndim * SST;        // S[t-2]^T
constexpr int Uo  = ST2 + Ndim * SST;        // x[t-3]@S[t-2]            (A) [64][136]
constexpr int V3o = Uo  + FIN * SA;          // U@S[t-1]                 (A)
constexpr int V2o = V3o + FIN * SA;          // x[t-2]@S[t-1]            (A)
constexpr int T1T = V2o + FIN * SA;          // tap1^T [128][72]         (B)
constexpr int T2T = T1T + Ndim * SBT;        // tap2^T
constexpr int T3T = T2T + Ndim * SBT;        // tap3^T
constexpr int LDS_ELEMS = T3T + Ndim * SBT;  // 141312 elems = 282624 B (< 320KB)
} // namespace gf

__device__ __forceinline__ v16bf load_frag16(const bf16_t* p0, const bf16_t* p1) {
  v8bf lo = *(const v8bf*)p0;
  v8bf hi = *(const v8bf*)p1;
  return __builtin_shufflevector(lo, hi, 0,1,2,3,4,5,6,7,8,9,10,11,12,13,14,15);
}

// ---- Staging helpers (float4 global reads) ----

// Row-major [64][SA] bf16 from global [64][128] fp32 (or zeros).
__device__ __forceinline__ void stage_rowmajor(bf16_t* lds, int base,
                                               const float* src, int tid) {
  using namespace gf;
  if (src) {
    const float4* s4 = (const float4*)src;
    for (int i = tid; i < (FIN * Ndim) / 4; i += 256) {
      const int f = i >> 5, n4 = (i & 31) * 4;
      float4 v = s4[i];
      v4bf p;
      p[0] = (bf16_t)v.x; p[1] = (bf16_t)v.y; p[2] = (bf16_t)v.z; p[3] = (bf16_t)v.w;
      *(v4bf*)(lds + base + f * SA + n4) = p;   // 8B ds_store_b64
    }
  } else {
    const v4bf z = {};
    for (int i = tid; i < (FIN * Ndim) / 4; i += 256) {
      const int f = i >> 5, n4 = (i & 31) * 4;
      *(v4bf*)(lds + base + f * SA + n4) = z;
    }
  }
}

// Transposed [cols][stride] bf16 from global [rows][cols] fp32.
__device__ __forceinline__ void stage_transposed(bf16_t* lds, int base, int stride,
                                                 const float* src, int rows, int cols,
                                                 int tid) {
  const float4* s4 = (const float4*)src;
  const int chunks_per_row = cols / 4;
  for (int i = tid; i < (rows * cols) / 4; i += 256) {
    const int r = i / chunks_per_row;
    const int c4 = (i - r * chunks_per_row) * 4;
    float4 v = s4[i];
    lds[base + (c4 + 0) * stride + r] = (bf16_t)v.x;
    lds[base + (c4 + 1) * stride + r] = (bf16_t)v.y;
    lds[base + (c4 + 2) * stride + r] = (bf16_t)v.z;
    lds[base + (c4 + 3) * stride + r] = (bf16_t)v.w;
  }
}

// C[64x128] = A[64x128] (row-major, stride SA) @ B[128x128] (stored transposed, stride SST).
// Wave handles tiles (mt, nt0..nt0+3). transC: write C transposed (stride strideC) else row-major.
__device__ __forceinline__ void mm64(bf16_t* lds, int Aoff, int Boff, int Coff,
                                     bool transC, int strideC,
                                     int lane, int mt, int nt0) {
  using namespace gf;
  const int m    = lane & 15;
  const int half = lane >> 4;
  const int kbA  = half * 8;    // A fragment K base within 32-chunk
  const int kbB  = half * 8;    // B fragment K base within 32-chunk
  v8f acc[4] = {};
#pragma unroll
  for (int kc = 0; kc < 4; ++kc) {
    const bf16_t* arow = lds + Aoff + (mt * 16 + m) * SA + kc * 32;
    v16bf a = load_frag16(arow + kbA, arow + 16 + kbA);
    v16bf bb[4];
#pragma unroll
    for (int nti = 0; nti < 4; ++nti) {
      const bf16_t* brow = lds + Boff + ((nt0 + nti) * 16 + m) * SST + kc * 32 + kbB;
      bb[nti] = load_frag16(brow, brow + 16);
    }
#pragma unroll
    for (int nti = 0; nti < 4; ++nti)
      acc[nti] = __builtin_amdgcn_wmma_f32_16x16x32_bf16(
          false, a, false, bb[nti], (short)0, acc[nti], false, false);
  }
#pragma unroll
  for (int nti = 0; nti < 4; ++nti) {
    const int n = (nt0 + nti) * 16 + m;
    if (transC) {
      v8bf cv;
#pragma unroll
      for (int r = 0; r < 8; ++r) cv[r] = (bf16_t)acc[nti][r];
      *(v8bf*)(lds + Coff + n * strideC + mt * 16 + half * 8) = cv;  // 16B ds_store_b128
    } else {
#pragma unroll
      for (int r = 0; r < 8; ++r)
        lds[Coff + (mt * 16 + half * 8 + r) * strideC + n] = (bf16_t)acc[nti][r];
    }
  }
}

__global__ void graph_filter_kernel(const float* __restrict__ x,
                                    const float* __restrict__ S,
                                    const float* __restrict__ H,
                                    const float* __restrict__ bias,
                                    float* __restrict__ out) {
  using namespace gf;
  extern __shared__ v8bf lds_v[];           // v8bf guarantees 16B base alignment
  bf16_t* lds = (bf16_t*)lds_v;

  const int bt   = blockIdx.x;
  const int b    = bt / Tdim;
  const int t    = bt - b * Tdim;
  const int tid  = threadIdx.x;
  const int lane = tid & 31;
  const int w    = tid >> 5;
  const int mt   = w & 3;            // output row tile (o)
  const int nt0  = (w >> 2) * 4;     // first output col tile (n)
  const int m    = lane & 15;
  const int half = lane >> 4;

  // ---- Stage x[t] transposed (B operand for tap0) ----
  stage_transposed(lds, X0T, SBT, x + (size_t)bt * FIN * Ndim, FIN, Ndim, tid);
  // ---- Stage x[t-1..t-3] row-major (A operands), zeros past the start ----
  stage_rowmajor(lds, X1o, (t >= 1) ? x + (size_t)(b * Tdim + t - 1) * FIN * Ndim : nullptr, tid);
  stage_rowmajor(lds, X2o, (t >= 2) ? x + (size_t)(b * Tdim + t - 2) * FIN * Ndim : nullptr, tid);
  stage_rowmajor(lds, X3o, (t >= 3) ? x + (size_t)(b * Tdim + t - 3) * FIN * Ndim : nullptr, tid);

  // ---- Stage S[t-d]^T for e = 0 (clamped indices only ever multiply zero taps) ----
#pragma unroll
  for (int d = 0; d < 3; ++d) {
    int ts = t - d; if (ts < 0) ts = 0;
    const float* sp = S + (((size_t)(b * Tdim + ts) * Edim + 0) * Ndim) * Ndim;
    stage_transposed(lds, ST0 + d * Ndim * SST, SST, sp, Ndim, Ndim, tid);
  }
  __syncthreads();

  v8f yacc[4] = {};

  for (int e = 0; e < Edim; ++e) {
    // ---- Diffusion chain (all WMMA) ----
    mm64(lds, X3o, ST2, Uo,  false, SA,  lane, mt, nt0);   // U  = x[t-3] @ S[t-2]
    __syncthreads();
    mm64(lds, Uo,  ST1, V3o, false, SA,  lane, mt, nt0);   // V3 = U @ S[t-1]
    mm64(lds, X2o, ST1, V2o, false, SA,  lane, mt, nt0);   // V2 = x[t-2] @ S[t-1]
    __syncthreads();
    mm64(lds, V3o, ST0, T3T, true,  SBT, lane, mt, nt0);   // tap3^T
    mm64(lds, V2o, ST0, T2T, true,  SBT, lane, mt, nt0);   // tap2^T
    mm64(lds, X1o, ST0, T1T, true,  SBT, lane, mt, nt0);   // tap1^T
    __syncthreads();

    // ---- Software pipeline: stage next-e S while projecting (St reads all done) ----
    if (e + 1 < Edim) {
#pragma unroll
      for (int d = 0; d < 3; ++d) {
        int ts = t - d; if (ts < 0) ts = 0;
        const float* sp = S + (((size_t)(b * Tdim + ts) * Edim + (e + 1)) * Ndim) * Ndim;
        stage_transposed(lds, ST0 + d * Ndim * SST, SST, sp, Ndim, Ndim, tid);
      }
    }

    // ---- Projection: yacc += H[:, e, k, :] @ tap_k  (tap0 = x[t]) ----
    const int kbA = half * 8;
    const int kbB = half * 8;
#pragma unroll
    for (int k = 0; k < Ktap; ++k) {
      const int Boff = (k == 0) ? X0T : (k == 1) ? T1T : (k == 2) ? T2T : T3T;
#pragma unroll
      for (int kc = 0; kc < 2; ++kc) {
        // H fragment straight from global (128KB total, L2-resident)
        const float* hp = H + (size_t)(mt * 16 + m) * HROW + e * (Ktap * FIN) + k * FIN + kc * 32;
        v16bf a;
#pragma unroll
        for (int j = 0; j < 8; ++j) {
          a[j]     = (bf16_t)hp[kbA + j];
          a[8 + j] = (bf16_t)hp[16 + kbA + j];
        }
        v16bf bb[4];
#pragma unroll
        for (int nti = 0; nti < 4; ++nti) {
          const bf16_t* brow = lds + Boff + ((nt0 + nti) * 16 + m) * SBT + kc * 32 + kbB;
          bb[nti] = load_frag16(brow, brow + 16);
        }
#pragma unroll
        for (int nti = 0; nti < 4; ++nti)
          yacc[nti] = __builtin_amdgcn_wmma_f32_16x16x32_bf16(
              false, a, false, bb[nti], (short)0, yacc[nti], false, false);
      }
    }
    __syncthreads();   // staged S visible + tap buffers free before next e
  }

  // ---- Epilogue: add bias, store fp32 ----
  float bv[8];
#pragma unroll
  for (int r = 0; r < 8; ++r) bv[r] = bias[mt * 16 + half * 8 + r];
  const size_t obase = (size_t)bt * FOUT * Ndim;
#pragma unroll
  for (int nti = 0; nti < 4; ++nti) {
    const int n = (nt0 + nti) * 16 + m;
#pragma unroll
    for (int r = 0; r < 8; ++r) {
      const int o = mt * 16 + half * 8 + r;
      out[obase + (size_t)o * Ndim + n] = yacc[nti][r] + bv[r];
    }
  }
}

extern "C" void kernel_launch(void* const* d_in, const int* in_sizes, int n_in,
                              void* d_out, int out_size, void* d_ws, size_t ws_size,
                              hipStream_t stream) {
  using namespace gf;
  const float* x    = (const float*)d_in[0];
  const float* S    = (const float*)d_in[1];
  const float* H    = (const float*)d_in[2];
  const float* bias = (const float*)d_in[3];
  float* out = (float*)d_out;

  const int Bdim = in_sizes[0] / (Tdim * FIN * Ndim);   // 16
  const size_t smem = (size_t)LDS_ELEMS * sizeof(bf16_t);

  hipFuncSetAttribute((const void*)graph_filter_kernel,
                      hipFuncAttributeMaxDynamicSharedMemorySize, (int)smem);

  dim3 grid(Bdim * Tdim);
  dim3 block(256);
  graph_filter_kernel<<<grid, block, smem, stream>>>(x, S, H, bias, out);
}
